// GeneratorBlock_12764642803916
// MI455X (gfx1250) — matrix operand order repaired and checked by reading, not compiled, not run on hardware
//
#include <hip/hip_runtime.h>

typedef __attribute__((ext_vector_type(16))) _Float16 v16h;
typedef __attribute__((ext_vector_type(8)))  _Float16 v8h;
typedef __attribute__((ext_vector_type(8)))  float    v8f;

static constexpr int   BN = 8;
static constexpr int   CI1 = 256, CO1 = 256, CI2 = 256, CO2 = 128;
static constexpr int   H1 = 64, W1 = 64, H2 = 128, W2 = 128;
static constexpr int   WD = 512;                  // w / n latent dims
static constexpr float EPSV = 1e-5f;
static constexpr float NEG_SLOPE = 0.1f;
static constexpr float RES_SCALE = 0.70710678118654752440f;

__device__ __forceinline__ float lrelu(float v) { return v >= 0.f ? v : NEG_SLOPE * v; }

// Async DMA: global -> LDS, 16 bytes per lane, tracked by ASYNCcnt.
// VGLOBAL async encoding: LDS byte offset in the VDST VGPR; global addr in VGPR pair.
__device__ __forceinline__ void async_g2l_b128(unsigned lds_byte_off, const void* g)
{
    unsigned long long ga = (unsigned long long)(uintptr_t)g;
    asm volatile("global_load_async_to_lds_b128 %0, %1, off"
                 :: "v"(lds_byte_off), "v"(ga)
                 : "memory");
}
__device__ __forceinline__ void wait_asynccnt0()
{
    asm volatile("s_wait_asynccnt 0" ::: "memory");
}

// ---------------------------------------------------------------------------
// K0: style / noise affines:  s = w @ A^T + b  (four small GEMVs, K=512)
// ---------------------------------------------------------------------------
__global__ __launch_bounds__(256) void affine_kernel(
    const float* __restrict__ w, const float* __restrict__ n,
    const float* __restrict__ A1w, const float* __restrict__ A1b,
    const float* __restrict__ B1w, const float* __restrict__ B1b,
    const float* __restrict__ A2w, const float* __restrict__ A2b,
    const float* __restrict__ B2w, const float* __restrict__ B2b,
    float* __restrict__ s1, float* __restrict__ n1,
    float* __restrict__ s2, float* __restrict__ n2)
{
    int tid = blockIdx.x * blockDim.x + threadIdx.x;
    int total = BN * (256 + 256 + 256 + 128);
    if (tid >= total) return;

    if (tid < BN * 256) {
        int b = tid / 256, o = tid % 256;
        float acc = A1b[o];
        for (int k = 0; k < WD; ++k) acc += w[b * WD + k] * A1w[o * WD + k];
        s1[b * 256 + o] = acc;
    } else if (tid < 2 * BN * 256) {
        int t = tid - BN * 256; int b = t / 256, o = t % 256;
        float acc = B1b[o];
        for (int k = 0; k < WD; ++k) acc += n[b * WD + k] * B1w[o * WD + k];
        n1[b * 256 + o] = acc;
    } else if (tid < 3 * BN * 256) {
        int t = tid - 2 * BN * 256; int b = t / 256, o = t % 256;
        float acc = A2b[o];
        for (int k = 0; k < WD; ++k) acc += w[b * WD + k] * A2w[o * WD + k];
        s2[b * 256 + o] = acc;
    } else {
        int t = tid - 3 * BN * 256; int b = t / 128, o = t % 128;
        float acc = B2b[o];
        for (int k = 0; k < WD; ++k) acc += n[b * WD + k] * B2w[o * WD + k];
        n2[b * 128 + o] = acc;
    }
}

// ---------------------------------------------------------------------------
// K1: demod[b,o] = rsqrt( sum_i s[b,i]^2 * sum_t w[o,i,t]^2 + eps )
// ---------------------------------------------------------------------------
__global__ __launch_bounds__(256) void demod_kernel(
    const float* __restrict__ s1, const float* __restrict__ s2,
    const float* __restrict__ w1, const float* __restrict__ w2,
    float* __restrict__ d1, float* __restrict__ d2)
{
    int tid = blockIdx.x * blockDim.x + threadIdx.x;
    if (tid < BN * CO1) {
        int b = tid / CO1, o = tid % CO1;
        float acc = 0.f;
        for (int i = 0; i < CI1; ++i) {
            const float* p = w1 + ((size_t)o * CI1 + i) * 9;
            float ws = 0.f;
            #pragma unroll
            for (int t = 0; t < 9; ++t) ws += p[t] * p[t];
            float s = s1[b * CI1 + i];
            acc += s * s * ws;
        }
        d1[tid] = rsqrtf(acc + EPSV);
    } else if (tid < BN * CO1 + BN * CO2) {
        int t0 = tid - BN * CO1; int b = t0 / CO2, o = t0 % CO2;
        float acc = 0.f;
        for (int i = 0; i < CI2; ++i) {
            const float* p = w2 + ((size_t)o * CI2 + i) * 9;
            float ws = 0.f;
            #pragma unroll
            for (int t = 0; t < 9; ++t) ws += p[t] * p[t];
            float s = s2[b * CI2 + i];
            acc += s * s * ws;
        }
        d2[t0] = rsqrtf(acc + EPSV);
    }
}

// ---------------------------------------------------------------------------
// K2: weight swizzle fp32 OIHW -> f16 WMMA A operand in contiguous k order (same as B):
// lane<16 holds K {0..15}, lane>=16 holds K {16..31}.
// Layout: [cotile][chunk][tap][lane][16 halfs] -> one 32B load per lane.
// ---------------------------------------------------------------------------
__global__ __launch_bounds__(256) void wswz_kernel(
    const float* __restrict__ wsrc, _Float16* __restrict__ dst,
    int cotiles, int chunks, int ci)
{
    int tid = blockIdx.x * blockDim.x + threadIdx.x;
    int total = cotiles * chunks * 9 * 32;
    if (tid >= total) return;
    int lane = tid & 31;
    int r = tid >> 5;
    int tap = r % 9; r /= 9;
    int chunk = r % chunks;
    int cotile = r / chunks;
    int co = cotile * 16 + (lane & 15);
    int klo = (lane >> 4) * 8;
    _Float16 vals[16];
    #pragma unroll
    for (int j = 0; j < 16; ++j) {
        int K = klo * 2 + j;
        int cich = chunk * 32 + K;
        vals[j] = (_Float16)wsrc[((size_t)co * ci + cich) * 9 + tap];
    }
    *(v16h*)(dst + (size_t)tid * 16) = *(v16h*)vals;
}

// ---------------------------------------------------------------------------
// K3: x (fp32 NCHW) * styles1 -> f16 NHWC  (B-operand layout for conv1)
// ---------------------------------------------------------------------------
__global__ __launch_bounds__(256) void prep1_kernel(
    const float* __restrict__ x, const float* __restrict__ s1,
    _Float16* __restrict__ xh)
{
    int tid = blockIdx.x * blockDim.x + threadIdx.x;          // BN*32*4096
    int hw = tid & 4095;
    int r = tid >> 12;
    int cig = r & 31;
    int b = r >> 5;
    if (b >= BN) return;
    v8h out;
    #pragma unroll
    for (int k = 0; k < 8; ++k) {
        int ci = cig * 8 + k;
        float v = x[(((size_t)b * CI1 + ci) << 12) + hw] * s1[b * CI1 + ci];
        out[k] = (_Float16)v;
    }
    *(v8h*)(xh + (((size_t)b << 12) + hw) * CI1 + cig * 8) = out;
}

// ---------------------------------------------------------------------------
// K4/K6: implicit-GEMM modulated conv3x3 via V_WMMA_F32_16X16X32_F16 with
// async global->LDS staging of the input halo strip.
//
// Workgroup = 4 waves; wave wv owns row h = bx*4+wv, one 16-channel cotile,
// full row of W pixels (PXT 16x16 tiles). Per Ci-chunk of 32:
//   - stage rows [h0-1 .. h0+4] x W x 32ch into LDS with
//     global_load_async_to_lds_b128 (zero-fill OOB rows via ds_store),
//   - s_wait_asynccnt 0 + barrier,
//   - 9 taps x PXT WMMAs; B operand = contiguous 32B LDS read per lane in
//     contiguous k order, same as A (lane<16: K 0..15 @ col N=lane; lane>=16:
//     K 16..31). LDS layout [row][Khalf][col][16 halfs].
// Fused epilogue: *demod + noise, LeakyReLU, f16 NHWC (+ optional f32 NCHW).
// ---------------------------------------------------------------------------
template <int H, int W, int CI, int CO, int PXT, bool STORE_F32>
__global__ __launch_bounds__(128) void modconv_wmma(
    const _Float16* __restrict__ Xin,   // NHWC f16, styles pre-applied
    const _Float16* __restrict__ Aw,    // swizzled f16 weights
    const float* __restrict__ demod,    // [BN, CO]
    const float* __restrict__ noise,    // [BN, CO]
    _Float16* __restrict__ Yh,          // NHWC f16 out
    float* __restrict__ Yf)             // NCHW f32 out (used if STORE_F32)
{
    constexpr int CHUNKS = CI / 32;
    constexpr int LGW = (W == 64) ? 6 : 7;
    constexpr int TILE_HALFS = 6 * 2 * W * 16;       // rows x Khalf x col x 16
    constexpr int NTRANS = TILE_HALFS / 8;           // b128 transfers (8 halfs)
    constexpr int PER_THREAD = NTRANS / 128;

    __shared__ _Float16 tile[TILE_HALFS];

    const int lane = threadIdx.x & 31;
    const int wv   = threadIdx.x >> 5;
    const int h0     = blockIdx.x * 4;
    const int h      = h0 + wv;
    const int cotile = blockIdx.y;
    const int b      = blockIdx.z;
    const int lmod = lane & 15;
    const int lhi  = lane >> 4;

    // C/D layout: VGPR r -> M = lhi*8 + r, N = lmod
    const int co_base = cotile * 16 + lhi * 8;
    float dm[8], ns[8];
    #pragma unroll
    for (int r = 0; r < 8; ++r) {
        dm[r] = demod[b * CO + co_base + r];
        ns[r] = noise[b * CO + co_base + r];
    }

    v8f acc[PXT] = {};

    const _Float16* xb = Xin + (size_t)b * H * W * CI;
    const _Float16* awbase = Aw + (size_t)cotile * CHUNKS * 9 * 32 * 16;

    // Per-thread transfer descriptors (chunk-invariant parts)
    int t_row[PER_THREAD], t_lh[PER_THREAD], t_col[PER_THREAD], t_j[PER_THREAD];
    #pragma unroll
    for (int i = 0; i < PER_THREAD; ++i) {
        int idx = threadIdx.x + i * 128;
        int j = idx & 1;
        int rem = idx >> 1;
        int col = rem & (W - 1);
        rem >>= LGW;
        int lh = rem & 1;
        int row = rem >> 1;                           // 0..5
        t_row[i] = row; t_lh[i] = lh; t_col[i] = col; t_j[i] = j;
    }

    for (int chunk = 0; chunk < CHUNKS; ++chunk) {
        __syncthreads();   // previous tile fully consumed before overwrite
        #pragma unroll
        for (int i = 0; i < PER_THREAD; ++i) {
            const int row = t_row[i], lh = t_lh[i], col = t_col[i], j = t_j[i];
            const int hy = h0 - 1 + row;
            const unsigned loff =
                (unsigned)((((row * 2 + lh) * W + col) * 16 + j * 8) * 2);
            if (hy >= 0 && hy < H) {
                async_g2l_b128(
                    (unsigned)(uintptr_t)((char*)tile + loff),
                    xb + ((size_t)hy * W + col) * CI + chunk * 32 + lh * 16 + j * 8);
            } else {
                v8h z = {};
                *(v8h*)((char*)tile + loff) = z;      // zero halo row
            }
        }
        wait_asynccnt0();
        __syncthreads();

        if (chunk + 1 < CHUNKS)
            __builtin_prefetch(awbase + ((size_t)(chunk + 1) * 9) * 32 * 16 + lane * 16, 0, 1);

        for (int tap = 0; tap < 9; ++tap) {
            const int dy = tap / 3 - 1, dx = tap % 3 - 1;
            v16h a = *(const v16h*)(awbase + (((size_t)chunk * 9 + tap) * 32 + lane) * 16);
            const int trow = wv + dy + 1;             // 0..5, always in-tile
            const _Float16* lrow = &tile[((trow * 2 + lhi) * W) * 16];
            #pragma unroll
            for (int t = 0; t < PXT; ++t) {
                int xx = t * 16 + lmod + dx;
                v16h bv = {};
                if (xx >= 0 && xx < W)
                    bv = *(const v16h*)(lrow + (size_t)xx * 16);
                acc[t] = __builtin_amdgcn_wmma_f32_16x16x32_f16(
                    false, a, false, bv, (short)0, acc[t], false, false);
            }
        }
    }

    #pragma unroll
    for (int t = 0; t < PXT; ++t) {
        int px = t * 16 + lmod;
        v8h oh;
        #pragma unroll
        for (int r = 0; r < 8; ++r) {
            float v = lrelu(acc[t][r] * dm[r] + ns[r]);
            oh[r] = (_Float16)v;
            if (STORE_F32)
                Yf[(((size_t)b * CO + co_base + r) * H + h) * W + px] = v;
        }
        *(v8h*)(Yh + (((size_t)b * H + h) * W + px) * CO + co_base) = oh;
    }
}

// ---------------------------------------------------------------------------
// K5: bilinear 2x upsample (half-pixel, edge clamp) of x1 (NHWC f16),
// fused with styles2 scaling, f16 NHWC out (B-operand layout for conv2)
// ---------------------------------------------------------------------------
__global__ __launch_bounds__(256) void upsample_scale_kernel(
    const _Float16* __restrict__ x1h, const float* __restrict__ s2,
    _Float16* __restrict__ x2in)
{
    int tid = blockIdx.x * blockDim.x + threadIdx.x;  // BN*128*128*32
    int cig = tid & 31;
    int r = tid >> 5;
    int ux = r & 127; r >>= 7;
    int uy = r & 127;
    int b = r >> 7;
    if (b >= BN) return;

    float fy = uy * 0.5f - 0.25f;
    float fx = ux * 0.5f - 0.25f;
    int y0 = (int)floorf(fy); float wy = fy - (float)y0;
    int x0 = (int)floorf(fx); float wx = fx - (float)x0;
    int y0c = y0 < 0 ? 0 : y0;          int y1c = (y0 + 1 > 63) ? 63 : y0 + 1;
    int x0c = x0 < 0 ? 0 : x0;          int x1c = (x0 + 1 > 63) ? 63 : x0 + 1;

    const _Float16* base = x1h + (size_t)b * 64 * 64 * CO1 + cig * 8;
    v8h p00 = *(const v8h*)(base + ((size_t)y0c * 64 + x0c) * CO1);
    v8h p01 = *(const v8h*)(base + ((size_t)y0c * 64 + x1c) * CO1);
    v8h p10 = *(const v8h*)(base + ((size_t)y1c * 64 + x0c) * CO1);
    v8h p11 = *(const v8h*)(base + ((size_t)y1c * 64 + x1c) * CO1);

    v8h o;
    #pragma unroll
    for (int k = 0; k < 8; ++k) {
        float v = (1.f - wy) * ((1.f - wx) * (float)p00[k] + wx * (float)p01[k])
                +         wy * ((1.f - wx) * (float)p10[k] + wx * (float)p11[k]);
        v *= s2[b * 256 + cig * 8 + k];
        o[k] = (_Float16)v;
    }
    *(v8h*)(x2in + (((size_t)b * 128 + uy) * 128 + ux) * CI2 + cig * 8) = o;
}

// ---------------------------------------------------------------------------
// K7: toRGB 1x1 conv (128->3) + bias + bilinear-upsampled rgb skip, *sqrt(.5)
// ---------------------------------------------------------------------------
__global__ __launch_bounds__(256) void torgb_kernel(
    const _Float16* __restrict__ x2h, const float* __restrict__ rgbw,
    const float* __restrict__ rgbb, const float* __restrict__ rgb,
    float* __restrict__ out)
{
    int tid = blockIdx.x * blockDim.x + threadIdx.x;   // BN*128*128
    if (tid >= BN * 128 * 128) return;
    int ux = tid & 127;
    int uy = (tid >> 7) & 127;
    int b  = tid >> 14;

    const _Float16* xp = x2h + (size_t)tid * CO2;
    float a0 = 0.f, a1 = 0.f, a2 = 0.f;
    for (int cc = 0; cc < CO2 / 16; ++cc) {
        v16h xv = *(const v16h*)(xp + cc * 16);
        #pragma unroll
        for (int j = 0; j < 16; ++j) {
            float xf = (float)xv[j];
            int co = cc * 16 + j;
            a0 += xf * rgbw[co];
            a1 += xf * rgbw[128 + co];
            a2 += xf * rgbw[256 + co];
        }
    }

    float fy = uy * 0.5f - 0.25f;
    float fx = ux * 0.5f - 0.25f;
    int y0 = (int)floorf(fy); float wy = fy - (float)y0;
    int x0 = (int)floorf(fx); float wx = fx - (float)x0;
    int y0c = y0 < 0 ? 0 : y0;          int y1c = (y0 + 1 > 63) ? 63 : y0 + 1;
    int x0c = x0 < 0 ? 0 : x0;          int x1c = (x0 + 1 > 63) ? 63 : x0 + 1;

    const size_t XOFF = (size_t)BN * CO2 * H2 * W2;   // fp32 x2 comes first
    float accs[3] = {a0, a1, a2};
    #pragma unroll
    for (int c = 0; c < 3; ++c) {
        const float* rb = rgb + (size_t)(b * 3 + c) * 4096;
        float up = (1.f - wy) * ((1.f - wx) * rb[y0c * 64 + x0c] + wx * rb[y0c * 64 + x1c])
                 +         wy * ((1.f - wx) * rb[y1c * 64 + x0c] + wx * rb[y1c * 64 + x1c]);
        out[XOFF + (size_t)(b * 3 + c) * 16384 + uy * 128 + ux] =
            (accs[c] + rgbb[c] + up) * RES_SCALE;
    }
}

// ---------------------------------------------------------------------------
extern "C" void kernel_launch(void* const* d_in, const int* in_sizes, int n_in,
                              void* d_out, int out_size, void* d_ws, size_t ws_size,
                              hipStream_t stream)
{
    const float* x    = (const float*)d_in[0];
    const float* w    = (const float*)d_in[1];
    const float* n    = (const float*)d_in[2];
    const float* rgb  = (const float*)d_in[3];
    const float* c1w  = (const float*)d_in[4];
    const float* A1w  = (const float*)d_in[5];
    const float* A1b  = (const float*)d_in[6];
    const float* B1w  = (const float*)d_in[7];
    const float* B1b  = (const float*)d_in[8];
    const float* c2w  = (const float*)d_in[9];
    const float* A2w  = (const float*)d_in[10];
    const float* A2b  = (const float*)d_in[11];
    const float* B2w  = (const float*)d_in[12];
    const float* B2b  = (const float*)d_in[13];
    const float* rgbw = (const float*)d_in[14];
    const float* rgbb = (const float*)d_in[15];

    char* ws = (char*)d_ws;
    // small fp32 vectors
    float* s1 = (float*)(ws + 0);          // 8x256
    float* n1 = (float*)(ws + 8192);       // 8x256
    float* s2 = (float*)(ws + 16384);      // 8x256
    float* n2 = (float*)(ws + 24576);      // 8x128
    float* d1 = (float*)(ws + 28672);      // 8x256
    float* d2 = (float*)(ws + 36864);      // 8x128
    // f16 regions (x2h aliases the dead x1in+x1h region)
    _Float16* Aw1  = (_Float16*)(ws + 40960);      // 1.18MB
    _Float16* Aw2  = (_Float16*)(ws + 1220608);    // 0.59MB
    _Float16* x1in = (_Float16*)(ws + 1810432);    // 16.78MB
    _Float16* x1h  = (_Float16*)(ws + 18587648);   // 16.78MB
    _Float16* x2h  = (_Float16*)(ws + 1810432);    // alias: 33.55MB
    _Float16* x2in = (_Float16*)(ws + 35364864);   // 67.1MB
    // total workspace use: ~102.5 MB

    float* out = (float*)d_out;

    affine_kernel<<<(BN * 896 + 255) / 256, 256, 0, stream>>>(
        w, n, A1w, A1b, B1w, B1b, A2w, A2b, B2w, B2b, s1, n1, s2, n2);

    demod_kernel<<<(BN * (CO1 + CO2) + 255) / 256, 256, 0, stream>>>(
        s1, s2, c1w, c2w, d1, d2);

    wswz_kernel<<<(16 * 8 * 9 * 32 + 255) / 256, 256, 0, stream>>>(
        c1w, Aw1, 16, 8, CI1);
    wswz_kernel<<<(8 * 8 * 9 * 32 + 255) / 256, 256, 0, stream>>>(
        c2w, Aw2, 8, 8, CI2);

    prep1_kernel<<<(BN * 32 * 4096) / 256, 256, 0, stream>>>(x, s1, x1in);

    {
        dim3 grid(H1 / 4, CO1 / 16, BN);
        modconv_wmma<H1, W1, CI1, CO1, 4, false>
            <<<grid, 128, 0, stream>>>(x1in, Aw1, d1, n1, x1h, nullptr);
    }

    upsample_scale_kernel<<<(BN * 128 * 128 * 32) / 256, 256, 0, stream>>>(
        x1h, s2, x2in);

    {
        dim3 grid(H2 / 4, CO2 / 16, BN);
        modconv_wmma<H2, W2, CI2, CO2, 8, true>
            <<<grid, 128, 0, stream>>>(x2in, Aw2, d2, n2, x2h, out);
    }

    torgb_kernel<<<(BN * 128 * 128 + 255) / 256, 256, 0, stream>>>(
        x2h, rgbw, rgbb, rgb, out);
}
